// EntNet_32332513804588
// MI455X (gfx1250) — matrix operand order repaired and checked by reading, not compiled, not run on hardware
//
#include <hip/hip_runtime.h>
#include <hip/hip_bf16.h>
#include <math.h>

// Problem constants (match reference)
#define B_   128
#define S_   64
#define L_   32
#define Q_   16
#define D_   256
#define NB_  20
#define NW_  32000

typedef __attribute__((ext_vector_type(16))) __bf16 bf16x16;
typedef __attribute__((ext_vector_type(8)))  float  f32x8;

union FragU { bf16x16 v; uint4 q[2]; };

// ---------------- workspace layout (bytes) ----------------
#define OFF_SALL   ((size_t)0)                         // B*S*D f32      = 8388608
#define OFF_SBF    ((size_t)8388608)                   // B*S*D bf16     = 4194304
#define OFF_SW     ((size_t)12582912)                  // B*S*D f32      = 8388608
#define OFF_Q      ((size_t)20971520)                  // B*D f32        = 131072
#define OFF_KVB    ((size_t)21102592)                  // NB*D f32       = 20480
#define OFF_UBF    ((size_t)21123072)                  // D*D bf16       = 131072
#define OFF_WBF    ((size_t)21254144)                  // D*D bf16       = 131072
#define OFF_RWBF   ((size_t)21385216)                  // NW*D bf16      = 16384000
#define OFF_YBF    ((size_t)37769216)                  // B*D bf16       = 65536

// ---------------- gather: s_all[b,s,d] = sum_l emb[inputs[b,s,l],d]*f_story[l,d]
__global__ __launch_bounds__(256) void gather_story(const int* __restrict__ inputs,
                                                    const float* __restrict__ emb,
                                                    const float* __restrict__ f_story,
                                                    float* __restrict__ sall,
                                                    __bf16* __restrict__ sbf) {
  int bs = blockIdx.x;            // b*S + s
  int d  = threadIdx.x;
  const int* row = inputs + (size_t)bs * L_;
  float acc = 0.f;
  for (int l = 0; l < L_; ++l) {
    int idx = row[l];
    acc += emb[(size_t)idx * D_ + d] * f_story[l * D_ + d];
  }
  sall[(size_t)bs * D_ + d] = acc;
  sbf [(size_t)bs * D_ + d] = (__bf16)acc;
}

// ---------------- q[b,d] = sum_j emb[query[b,j],d]*f_query[j,d]
__global__ __launch_bounds__(256) void gather_query(const int* __restrict__ query,
                                                    const float* __restrict__ emb,
                                                    const float* __restrict__ f_query,
                                                    float* __restrict__ qv) {
  int b = blockIdx.x;
  int d = threadIdx.x;
  float acc = 0.f;
  for (int j = 0; j < Q_; ++j) {
    int idx = query[b * Q_ + j];
    acc += emb[(size_t)idx * D_ + d] * f_query[j * D_ + d];
  }
  qv[(size_t)b * D_ + d] = acc;
}

// ---------------- kVb[n,e] = sum_d keys[n,d]*V[e,d] + bias[e]
__global__ __launch_bounds__(256) void kvb_kernel(const float* __restrict__ emb,
                                                  const float* __restrict__ V,
                                                  const float* __restrict__ bias,
                                                  float* __restrict__ kVb) {
  int e = threadIdx.x;
  const float* keys = emb + (size_t)NW_ * D_;
  for (int n = 0; n < NB_; ++n) {
    float acc = bias[e];
    const float* kr = keys + n * D_;
    const float* vr = V + e * D_;
    for (int d = 0; d < D_; ++d) acc += kr[d] * vr[d];
    kVb[n * D_ + e] = acc;
  }
}

// ---------------- f32 -> bf16 conversion (grid-stride)
__global__ __launch_bounds__(256) void cvt_bf16(const float* __restrict__ src,
                                                __bf16* __restrict__ dst, int n) {
  for (int i = blockIdx.x * 256 + threadIdx.x; i < n; i += gridDim.x * 256)
    dst[i] = (__bf16)src[i];
}

// ---------------- sW = s_bf16 (8192x256) @ Wbf^T   (B-operand row-major: B[k][e]=W[e][k])
__global__ __launch_bounds__(256) void sw_gemm(const __bf16* __restrict__ sbf,
                                               const __bf16* __restrict__ wbf,
                                               float* __restrict__ sW) {
  int wv = threadIdx.x >> 5, lane = threadIdx.x & 31;
  int rowbase = (blockIdx.x * 8 + wv) * 16;   // M tile
  int colbase = blockIdx.y * 16;              // N tile
  int arow = rowbase + (lane & 15);
  int kA   = (lane >> 4) * 8;
  int bcol = colbase + (lane & 15);
  int kB   = (lane >> 4) * 8;
  f32x8 c = {};
  for (int kc = 0; kc < 8; ++kc) {
    FragU a, b;
    const __bf16* ap = sbf + (size_t)arow * D_ + kc * 32 + kA;
    a.q[0] = *(const uint4*)(ap);
    a.q[1] = *(const uint4*)(ap + 16);
    const __bf16* bp = wbf + (size_t)bcol * D_ + kc * 32 + kB;
    b.q[0] = *(const uint4*)(bp);
    b.q[1] = *(const uint4*)(bp + 16);
    c = __builtin_amdgcn_wmma_f32_16x16x32_bf16(false, a.v, false, b.v, (short)0, c, false, false);
  }
  int r0 = rowbase + 8 * (lane >> 4);
  int col = colbase + (lane & 15);
  for (int r = 0; r < 8; ++r) sW[(size_t)(r0 + r) * D_ + col] = c[r];
}

// ---------------- recurrent scan, one workgroup per batch element
__global__ __launch_bounds__(256, 1) void scan_kernel(const float* __restrict__ emb,
                                                      const float* __restrict__ sall,
                                                      const float* __restrict__ sW,
                                                      const float* __restrict__ kVb_g,
                                                      const __bf16* __restrict__ ubf,
                                                      const float* __restrict__ qv_g,
                                                      const float* __restrict__ a_mem_p,
                                                      const float* __restrict__ a_out_p,
                                                      const float* __restrict__ Hw,
                                                      const float* __restrict__ Hb,
                                                      __bf16* __restrict__ ybf) {
  extern __shared__ char smem[];
  float*  hF   = (float*)(smem);                       // 32*256 f32   (rows>=NB stay 0)
  __bf16* hB   = (__bf16*)(smem + 32768);              // 32*256 bf16
  float*  keys = (float*)(smem + 49152);               // 20*256 f32
  float*  kVb  = (float*)(smem + 69632);               // 20*256 f32 (kV + bias)
  float*  sBuf = (float*)(smem + 90112);               // 256
  float*  sWr  = (float*)(smem + 91136);               // 256
  float*  gB   = (float*)(smem + 92160);               // 32
  float*  lse  = (float*)(smem + 92288);               // 32
  float*  uB   = (float*)(smem + 92416);               // 256
  __bf16* uLds = (__bf16*)(smem + 93440);              // 256*256 bf16 (U row-major)

  const int tid = threadIdx.x, b = blockIdx.x;
  const int wv = tid >> 5, lane = tid & 31;
  const float am = *a_mem_p;

  // Stage U into LDS with the CDNA5 async cache->LDS path (ASYNCcnt-tracked):
  // 8192 x 16B transfers, no VGPR round-trip; overlaps with the scalar init below.
  {
    unsigned lds_base = (unsigned)(uintptr_t)uLds;     // LDS byte address (flat addr low bits)
    for (int i = tid; i < (D_ * D_) / 8; i += 256) {
      unsigned long long gsrc = (unsigned long long)(uintptr_t)(((const uint4*)ubf) + i);
      unsigned loff = lds_base + (unsigned)i * 16u;
      asm volatile("global_load_async_to_lds_b128 %0, %1, off"
                   :: "v"(loff), "v"(gsrc) : "memory");
    }
  }

  // init: keys + kV, h0 = keys (rows >= NB zeroed)
  for (int i = tid; i < NB_ * D_; i += 256) {
    keys[i] = emb[(size_t)NW_ * D_ + i];
    kVb[i]  = kVb_g[i];
  }
  for (int i = tid; i < 32 * D_; i += 256) {
    float v = (i < NB_ * D_) ? emb[(size_t)NW_ * D_ + i] : 0.f;
    hF[i] = v;
    hB[i] = (__bf16)v;
  }
  asm volatile("s_wait_asynccnt 0x0" ::: "memory");    // drain async U staging
  __syncthreads();

  for (int t = 0; t < S_; ++t) {
    // stage s_t and precomputed (s W^T)_t
    const float* srow  = sall + ((size_t)b * S_ + t) * D_;
    const float* swrow = sW   + ((size_t)b * S_ + t) * D_;
    sBuf[tid] = srow[tid];
    sWr[tid]  = swrow[tid];
    __syncthreads();

    // gate: g[n] = sigmoid((h[n]+keys[n]) . s)    (wave-parallel rows)
    for (int rr = 0; rr < 3; ++rr) {
      int n = wv + 8 * rr;
      if (n < NB_) {
        float p = 0.f;
        for (int i = lane; i < D_; i += 32)
          p += (hF[n * D_ + i] + keys[n * D_ + i]) * sBuf[i];
        for (int off = 16; off; off >>= 1) p += __shfl_xor(p, off, 32);
        if (lane == 0) gB[n] = 1.f / (1.f + expf(-p));
      }
    }
    __syncthreads();

    // cand = prelu(h @ U^T + kV + sW + bias); h += g*cand   -- WMMA 2x16 tile grid
    for (int tt = 0; tt < 4; ++tt) {
      int tile = wv + 8 * tt;
      int mt = tile & 1, nt = tile >> 1;
      int arow = mt * 16 + (lane & 15);
      int kA   = (lane >> 4) * 8;
      int bcol = nt * 16 + (lane & 15);
      int kB   = (lane >> 4) * 8;
      f32x8 c = {};
      for (int kc = 0; kc < 8; ++kc) {
        FragU a, bb;
        const __bf16* ap = hB + arow * D_ + kc * 32 + kA;
        a.q[0] = *(const uint4*)(ap);
        a.q[1] = *(const uint4*)(ap + 16);
        const __bf16* bp = uLds + bcol * D_ + kc * 32 + kB;
        bb.q[0] = *(const uint4*)(bp);
        bb.q[1] = *(const uint4*)(bp + 16);
        c = __builtin_amdgcn_wmma_f32_16x16x32_bf16(false, a.v, false, bb.v, (short)0, c, false, false);
      }
      int e  = nt * 16 + (lane & 15);
      int m0 = mt * 16 + 8 * (lane >> 4);
      float addc = sWr[e];
      for (int r = 0; r < 8; ++r) {
        int m = m0 + r;
        if (m < NB_) {
          float cand = c[r] + kVb[m * D_ + e] + addc;
          cand = cand >= 0.f ? cand : am * cand;
          hF[m * D_ + e] += gB[m] * cand;
        }
      }
    }
    __syncthreads();

    // row-normalize, refresh bf16 copy
    for (int rr = 0; rr < 3; ++rr) {
      int n = wv + 8 * rr;
      if (n < NB_) {
        float ss = 0.f;
        for (int i = lane; i < D_; i += 32) { float v = hF[n * D_ + i]; ss += v * v; }
        for (int off = 16; off; off >>= 1) ss += __shfl_xor(ss, off, 32);
        float sc = rsqrtf(ss);
        for (int i = lane; i < D_; i += 32) {
          float v = hF[n * D_ + i] * sc;
          hF[n * D_ + i] = v;
          hB[n * D_ + i] = (__bf16)v;
        }
      }
    }
    __syncthreads();
  }

  // readout head: ls = log_softmax(q*h, d); u = sum_n ls*h; y = prelu(q + u@Hw^T + Hb)
  sBuf[tid] = qv_g[(size_t)b * D_ + tid];
  __syncthreads();
  for (int rr = 0; rr < 3; ++rr) {
    int n = wv + 8 * rr;
    if (n < NB_) {
      float mx = -1e30f;
      for (int i = lane; i < D_; i += 32) mx = fmaxf(mx, sBuf[i] * hF[n * D_ + i]);
      for (int off = 16; off; off >>= 1) mx = fmaxf(mx, __shfl_xor(mx, off, 32));
      float se = 0.f;
      for (int i = lane; i < D_; i += 32) se += expf(sBuf[i] * hF[n * D_ + i] - mx);
      for (int off = 16; off; off >>= 1) se += __shfl_xor(se, off, 32);
      if (lane == 0) lse[n] = mx + logf(se);
    }
  }
  __syncthreads();
  {
    int d = tid;
    float u = 0.f;
    for (int n = 0; n < NB_; ++n) {
      float h = hF[n * D_ + d];
      u += (sBuf[d] * h - lse[n]) * h;
    }
    uB[d] = u;
  }
  __syncthreads();
  {
    int e = tid;
    float acc = sBuf[e] + Hb[e];
    const float* hr = Hw + (size_t)e * D_;
    for (int d = 0; d < D_; ++d) acc += uB[d] * hr[d];
    float ao = *a_out_p;
    acc = acc >= 0.f ? acc : ao * acc;
    ybf[(size_t)b * D_ + e] = (__bf16)acc;
  }
}

// ---------------- out = y(128x256) @ Rw^T + Rb  (B[k][w] = Rw[w][k] row-major)
__global__ __launch_bounds__(256) void out_gemm(const __bf16* __restrict__ ybf,
                                                const __bf16* __restrict__ rwbf,
                                                const float* __restrict__ rb,
                                                float* __restrict__ out) {
  int wv = threadIdx.x >> 5, lane = threadIdx.x & 31;
  int rowbase = wv * 16;
  int colbase = blockIdx.x * 16;
  int arow = rowbase + (lane & 15);
  int kA   = (lane >> 4) * 8;
  int bcol = colbase + (lane & 15);
  int kB   = (lane >> 4) * 8;
  f32x8 c = {};
  for (int kc = 0; kc < 8; ++kc) {
    FragU a, b;
    const __bf16* ap = ybf + (size_t)arow * D_ + kc * 32 + kA;
    a.q[0] = *(const uint4*)(ap);
    a.q[1] = *(const uint4*)(ap + 16);
    const __bf16* bp = rwbf + (size_t)bcol * D_ + kc * 32 + kB;
    // pull next K-chunk of the streamed R_w panel ahead of the WMMA chain
    if (kc < 7) __builtin_prefetch(bp + 32, 0, 0);
    b.q[0] = *(const uint4*)(bp);
    b.q[1] = *(const uint4*)(bp + 16);
    c = __builtin_amdgcn_wmma_f32_16x16x32_bf16(false, a.v, false, b.v, (short)0, c, false, false);
  }
  int col = colbase + (lane & 15);
  float bias = rb[col];
  int r0 = rowbase + 8 * (lane >> 4);
  for (int r = 0; r < 8; ++r)
    out[(size_t)(r0 + r) * NW_ + col] = c[r] + bias;
}

// ---------------- host ----------------
extern "C" void kernel_launch(void* const* d_in, const int* in_sizes, int n_in,
                              void* d_out, int out_size, void* d_ws, size_t ws_size,
                              hipStream_t stream) {
  const int*   inputs  = (const int*)  d_in[0];
  const int*   query   = (const int*)  d_in[1];
  const float* emb     = (const float*)d_in[2];
  const float* f_story = (const float*)d_in[3];
  const float* f_query = (const float*)d_in[4];
  const float* U       = (const float*)d_in[5];
  const float* V       = (const float*)d_in[6];
  const float* W       = (const float*)d_in[7];
  const float* bias    = (const float*)d_in[8];
  const float* a_mem   = (const float*)d_in[9];
  const float* Hw      = (const float*)d_in[10];
  const float* Hb      = (const float*)d_in[11];
  const float* Rw      = (const float*)d_in[12];
  const float* Rb      = (const float*)d_in[13];
  const float* a_out   = (const float*)d_in[14];
  float* out = (float*)d_out;
  char*  ws  = (char*)d_ws;

  float*  sall = (float*) (ws + OFF_SALL);
  __bf16* sbf  = (__bf16*)(ws + OFF_SBF);
  float*  sW   = (float*) (ws + OFF_SW);
  float*  qv   = (float*) (ws + OFF_Q);
  float*  kVb  = (float*) (ws + OFF_KVB);
  __bf16* ubf  = (__bf16*)(ws + OFF_UBF);
  __bf16* wbf  = (__bf16*)(ws + OFF_WBF);
  __bf16* rwbf = (__bf16*)(ws + OFF_RWBF);
  __bf16* ybf  = (__bf16*)(ws + OFF_YBF);

  gather_story<<<B_ * S_, 256, 0, stream>>>(inputs, emb, f_story, sall, sbf);
  gather_query<<<B_, 256, 0, stream>>>(query, emb, f_query, qv);
  kvb_kernel<<<1, 256, 0, stream>>>(emb, V, bias, kVb);
  cvt_bf16<<<256, 256, 0, stream>>>(U, ubf, D_ * D_);
  cvt_bf16<<<256, 256, 0, stream>>>(W, wbf, D_ * D_);
  cvt_bf16<<<2048, 256, 0, stream>>>(Rw, rwbf, NW_ * D_);
  sw_gemm<<<dim3(64, 16), 256, 0, stream>>>(sbf, wbf, sW);
  scan_kernel<<<B_, 256, 224512, stream>>>(emb, sall, sW, kVb, ubf, qv,
                                           a_mem, a_out, Hw, Hb, ybf);
  out_gemm<<<NW_ / 16, 256, 0, stream>>>(ybf, rwbf, Rb, out);
}
